// FM4BioLayer_833223656397
// MI455X (gfx1250) — matrix operand order repaired and checked by reading, not compiled, not run on hardware
//
#include <hip/hip_runtime.h>
#include <hip/hip_bf16.h>

// ---------------------------------------------------------------------------
// MI455X / gfx1250 transformer layer (pre-LN attention + SwiGLU MLP)
// ~172 GFLOP vs ~85MB HBM traffic -> compute-bound -> bf16 WMMA everywhere.
// This round: GLOBAL_LOAD_ASYNC_TO_LDS_B128 (ASYNCcnt) for non-transposing
// tile staging + ping-pong double buffering so async DMA runs under WMMA.
// ---------------------------------------------------------------------------

typedef __attribute__((ext_vector_type(8)))  __bf16 v8bf;
typedef __attribute__((ext_vector_type(16))) __bf16 v16bf;
typedef __attribute__((ext_vector_type(8)))  float  v8f;

static __device__ __forceinline__ v8f wmma_bf16(v16bf a, v16bf b, v8f c) {
  // (neg_a, A, neg_b, B, c_mod, C, reuse_a, reuse_b)
  return __builtin_amdgcn_wmma_f32_16x16x32_bf16(false, a, false, b, (short)0, c,
                                                 false, false);
}

static __device__ __forceinline__ v8f v8f_zero() {
  v8f z = {0.f, 0.f, 0.f, 0.f, 0.f, 0.f, 0.f, 0.f};
  return z;
}

// --- CDNA5 async memory->LDS (ASYNCcnt-tracked, no VGPR transit) -----------
// LDS aperture flat addresses carry the LDS byte offset in addr[31:0].
static __device__ __forceinline__ void async_load_b128(const void* gptr,
                                                       void* lptr) {
  unsigned lds = (unsigned)(unsigned long long)lptr;
  unsigned long long g = (unsigned long long)gptr;
  asm volatile("global_load_async_to_lds_b128 %0, %1, off"
               :: "v"(lds), "v"(g)
               : "memory");
}
static __device__ __forceinline__ void wait_async() {
  asm volatile("s_wait_asynccnt 0x0" ::: "memory");
}

// A-matrix fragment (16-bit, 16x32, ISA 7.12.2): lane half khalf holds
// K = khalf*8 + [0..7]  and  K = 16 + khalf*8 + [0..7]    (interleaved 8s)
static __device__ __forceinline__ v16bf ld_frag_a(const __bf16* rowp, int khalf) {
  v8bf lo = *(const v8bf*)(rowp + khalf * 8);
  v8bf hi = *(const v8bf*)(rowp + khalf * 8 + 16);
  return __builtin_shufflevector(lo, hi, 0, 1, 2, 3, 4, 5, 6, 7,
                                 8, 9, 10, 11, 12, 13, 14, 15);
}

// B-matrix fragment (16-bit, 32x16): lane half khalf holds
// K = khalf*8 + [0..7]  and  K = 16 + khalf*8 + [0..7]  (same order as A)
static __device__ __forceinline__ v16bf ld_frag_b(const __bf16* rowp, int khalf) {
  v8bf lo = *(const v8bf*)(rowp + khalf * 8);
  v8bf hi = *(const v8bf*)(rowp + khalf * 8 + 16);
  return __builtin_shufflevector(lo, hi, 0, 1, 2, 3, 4, 5, 6, 7,
                                 8, 9, 10, 11, 12, 13, 14, 15);
}

// ---------------------------------------------------------------------------
// fp32 -> bf16 conversion (weights, one pass each call; deterministic)
// ---------------------------------------------------------------------------
__global__ void cvt_f32_bf16(const float* __restrict__ in,
                             __bf16* __restrict__ out, long long n) {
  long long i = (long long)blockIdx.x * blockDim.x + threadIdx.x;
  long long stride = (long long)gridDim.x * blockDim.x;
  for (; i < n; i += stride) out[i] = (__bf16)in[i];
}

// ---------------------------------------------------------------------------
// LayerNorm over D=1024, one row per block (256 threads = 8 waves)
// ---------------------------------------------------------------------------
__global__ void __launch_bounds__(256)
ln_kernel(const float* __restrict__ x, const float* __restrict__ g,
          const float* __restrict__ bta, __bf16* __restrict__ out, int D) {
  const int row = blockIdx.x;
  const float* xr = x + (size_t)row * D;
  float s = 0.f, s2 = 0.f;
  for (int i = threadIdx.x; i < D; i += 256) {
    float v = xr[i];
    s += v;
    s2 += v * v;
  }
#pragma unroll
  for (int off = 16; off >= 1; off >>= 1) {
    s += __shfl_xor(s, off, 32);
    s2 += __shfl_xor(s2, off, 32);
  }
  __shared__ float rs[8], rs2[8];
  const int wave = threadIdx.x >> 5, lane = threadIdx.x & 31;
  if (lane == 0) { rs[wave] = s; rs2[wave] = s2; }
  __syncthreads();
  s = 0.f; s2 = 0.f;
#pragma unroll
  for (int w = 0; w < 8; ++w) { s += rs[w]; s2 += rs2[w]; }
  const float mu = s / D;
  const float var = s2 / D - mu * mu;
  const float inv = rsqrtf(var + 1e-5f);
  for (int i = threadIdx.x; i < D; i += 256)
    out[(size_t)row * D + i] = (__bf16)((xr[i] - mu) * inv * g[i] + bta[i]);
}

// ---------------------------------------------------------------------------
// Tiled bf16 WMMA GEMM: C[M,N] = A[M,K] * W[K,N] + bias (+res), 128x128x32
// Double-buffered: A tile via async-to-LDS, W tile via transposed scatter.
// 8 waves, each wave: 64x32 patch = 4x2 grid of 16x16 WMMA tiles.
// ---------------------------------------------------------------------------
template <bool BF16OUT, bool RES>
__global__ void __launch_bounds__(256)
gemm_bf16_kernel(const __bf16* __restrict__ A, const __bf16* __restrict__ W,
                 const float* __restrict__ bias, const float* __restrict__ res,
                 float* __restrict__ Cf, __bf16* __restrict__ Cb,
                 int M, int N, int K) {
  __shared__ __align__(16) __bf16 As[2][128][32];  // A tile, row-major
  __shared__ __align__(16) __bf16 Bt[2][128][32];  // W tile transposed: Bt[n][k]

  const int tid = threadIdx.x;
  const int wave = tid >> 5, lane = tid & 31;
  const int wm = wave >> 2, wn = wave & 3;   // 2x4 wave grid
  const int lm = lane & 15, khalf = lane >> 4;
  const int block_m = blockIdx.y * 128;
  const int block_n = blockIdx.x * 128;

  v8f acc[4][2];
#pragma unroll
  for (int t = 0; t < 4; ++t)
#pragma unroll
    for (int u = 0; u < 2; ++u) acc[t][u] = v8f_zero();

  auto stage = [&](int buf, int k0) {
    // A tile 128x32 bf16: 512 x 16B chunks, 2 per thread, async to LDS
#pragma unroll
    for (int it = 0; it < 2; ++it) {
      int q = tid + it * 256;            // 0..511
      int r = q >> 2;                    // 0..127
      int s = (q & 3) * 8;               // 0,8,16,24
      async_load_b128(A + (size_t)(block_m + r) * K + k0 + s, &As[buf][r][s]);
    }
    // W tile 32x128 -> transposed scatter Bt[n][k] (sync; async can't transpose)
#pragma unroll
    for (int it = 0; it < 2; ++it) {
      int chunk = tid + it * 256;        // 512 chunks of 8 bf16
      int kk = chunk >> 4;               // 0..31
      int nn = (chunk & 15) * 8;         // 0..120
      union { uint4 u; __bf16 e[8]; } tb;
      tb.u = *(const uint4*)(W + (size_t)(k0 + kk) * N + block_n + nn);
#pragma unroll
      for (int j = 0; j < 8; ++j) Bt[buf][nn + j][kk] = tb.e[j];
    }
  };

  stage(0, 0);
  int buf = 0;
  for (int k0 = 0; k0 < K; k0 += 32, buf ^= 1) {
    wait_async();
    __syncthreads();
    if (k0 + 32 < K) {
      stage(buf ^ 1, k0 + 32);  // next tile streams in under the WMMAs
      __builtin_prefetch(W + (size_t)(k0 + 32) * N + block_n, 0, 1);
    }
    v16bf af[4], bf[2];
#pragma unroll
    for (int t = 0; t < 4; ++t)
      af[t] = ld_frag_a(&As[buf][wm * 64 + t * 16 + lm][0], khalf);
#pragma unroll
    for (int u = 0; u < 2; ++u)
      bf[u] = ld_frag_b(&Bt[buf][wn * 32 + u * 16 + lm][0], khalf);
#pragma unroll
    for (int t = 0; t < 4; ++t)
#pragma unroll
      for (int u = 0; u < 2; ++u)
        acc[t][u] = wmma_bf16(af[t], bf[u], acc[t][u]);
  }

  // epilogue: C layout -> row = r + (lane<16?0:8), col = lane%16
  const int mrow_off = (lane < 16) ? 0 : 8;
#pragma unroll
  for (int t = 0; t < 4; ++t) {
#pragma unroll
    for (int u = 0; u < 2; ++u) {
      const int n = block_n + wn * 32 + u * 16 + lm;
      const float bv = bias[n];
#pragma unroll
      for (int r = 0; r < 8; ++r) {
        const int m = block_m + wm * 64 + t * 16 + mrow_off + r;
        const size_t o = (size_t)m * N + n;
        float v = acc[t][u][r] + bv;
        if (RES) v += res[o];
        if (BF16OUT) Cb[o] = (__bf16)v;
        else Cf[o] = v;
      }
    }
  }
}

// ---------------------------------------------------------------------------
// RoPE on Q,K (full head-dim rotate) + repack Q,K,V to [B,H,S,64] bf16
// ---------------------------------------------------------------------------
__global__ void rope_pack_kernel(const float* __restrict__ Qf,
                                 const float* __restrict__ Kf,
                                 const float* __restrict__ Vf,
                                 const float* __restrict__ freqs,
                                 __bf16* __restrict__ Qb, __bf16* __restrict__ Kb,
                                 __bf16* __restrict__ Vb, int Bsz, int Hn,
                                 int Sn, int Dm) {
  int idx = blockIdx.x * blockDim.x + threadIdx.x;
  int tot = Bsz * Sn * Hn * 32;
  if (idx >= tot) return;
  int d = idx & 31;
  int t1 = idx >> 5;
  int h = t1 % Hn;
  t1 /= Hn;
  int s = t1 % Sn;
  int b = t1 / Sn;
  float f = freqs[s * 64 + d];  // freqs[:,d] == freqs[:,d+32] (concat)
  float c = cosf(f), sn = sinf(f);
  size_t src = ((size_t)(b * Sn + s)) * Dm + h * 64 + d;
  float q1 = Qf[src], q2 = Qf[src + 32];
  float k1 = Kf[src], k2 = Kf[src + 32];
  size_t dst = ((size_t)((b * Hn + h) * Sn + s)) * 64 + d;
  Qb[dst]      = (__bf16)(q1 * c - q2 * sn);
  Qb[dst + 32] = (__bf16)(q2 * c + q1 * sn);
  Kb[dst]      = (__bf16)(k1 * c - k2 * sn);
  Kb[dst + 32] = (__bf16)(k2 * c + k1 * sn);
  Vb[dst]      = (__bf16)Vf[src];
  Vb[dst + 32] = (__bf16)Vf[src + 32];
}

// ---------------------------------------------------------------------------
// Flash attention, one (b,h) x 128 q-rows per block. 8 waves x 16 q-rows.
// KV double-buffered: K chunk async-to-LDS (row-major == B-frag layout),
// V chunk transposed via sync scatter. Online softmax; WMMA QK^T and PV.
// ---------------------------------------------------------------------------
__global__ void __launch_bounds__(256)
attn_kernel(const __bf16* __restrict__ Qb, const __bf16* __restrict__ Kb,
            const __bf16* __restrict__ Vb, const float* __restrict__ mask,
            __bf16* __restrict__ ctx, int Hn, int Sn, int Dm) {
  __shared__ __align__(16) __bf16 Ks[2][32][64];   // K chunk (B-frag layout)
  __shared__ __align__(16) __bf16 Vt[2][64][32];   // V chunk transposed
  __shared__ __align__(16) __bf16 Pb[8][16][32];   // per-wave P relayout buffer

  const int tid = threadIdx.x;
  const int wave = tid >> 5, lane = tid & 31;
  const int lm = lane & 15, khalf = lane >> 4;
  const int bh = blockIdx.y;
  const int b = bh / Hn, h = bh % Hn;
  const int qrow0 = blockIdx.x * 128 + wave * 16;

  const __bf16* Qh = Qb + (size_t)bh * Sn * 64;
  const __bf16* Kh = Kb + (size_t)bh * Sn * 64;
  const __bf16* Vh = Vb + (size_t)bh * Sn * 64;

  // Q fragments (HD=64 -> 2 k-steps) pinned in registers for the whole loop
  v16bf qf[2];
#pragma unroll
  for (int kk = 0; kk < 2; ++kk)
    qf[kk] = ld_frag_a(Qh + (size_t)(qrow0 + lm) * 64 + kk * 32, khalf);

  v8f cacc[4];
#pragma unroll
  for (int t = 0; t < 4; ++t) cacc[t] = v8f_zero();
  float mrow[8], lrow[8], alpha[8];
#pragma unroll
  for (int r = 0; r < 8; ++r) { mrow[r] = -1e30f; lrow[r] = 0.f; }
  const float scale = 0.125f;  // 1/sqrt(64)

  auto stageKV = [&](int buf, int kv0) {
    int r = tid >> 3, c = (tid & 7) * 8;  // 32 rows x 64, 8 bf16 per thread
    async_load_b128(Kh + (size_t)(kv0 + r) * 64 + c, &Ks[buf][r][c]);
    union { uint4 u; __bf16 e[8]; } tb;
    tb.u = *(const uint4*)(Vh + (size_t)(kv0 + r) * 64 + c);
#pragma unroll
    for (int j = 0; j < 8; ++j) Vt[buf][c + j][r] = tb.e[j];
  };

  stageKV(0, 0);
  int buf = 0;
  for (int kv0 = 0; kv0 < Sn; kv0 += 32, buf ^= 1) {
    wait_async();
    __syncthreads();
    if (kv0 + 32 < Sn) stageKV(buf ^ 1, kv0 + 32);

    // scores: 16x32 = two 16x16 C tiles, K=64 = two WMMA k-steps each
    v8f sc[2];
    sc[0] = v8f_zero();
    sc[1] = v8f_zero();
#pragma unroll
    for (int u = 0; u < 2; ++u)
#pragma unroll
      for (int kk = 0; kk < 2; ++kk)
        sc[u] = wmma_bf16(qf[kk],
                          ld_frag_b(&Ks[buf][u * 16 + lm][kk * 32], khalf),
                          sc[u]);

    float mv[2];
    bool kill[2];
#pragma unroll
    for (int u = 0; u < 2; ++u) {
      mv[u] = mask[(size_t)b * Sn + kv0 + u * 16 + lm];
      kill[u] = mv[u] < -1e-5f;
#pragma unroll
      for (int r = 0; r < 8; ++r) sc[u][r] = sc[u][r] * scale + mv[u];
    }
    // online softmax: row reductions across each 16-lane half
#pragma unroll
    for (int r = 0; r < 8; ++r) {
      float v = fmaxf(sc[0][r], sc[1][r]);
#pragma unroll
      for (int m = 8; m >= 1; m >>= 1) v = fmaxf(v, __shfl_xor(v, m, 16));
      float mnew = fmaxf(mrow[r], v);
      alpha[r] = __expf(mrow[r] - mnew);
      mrow[r] = mnew;
    }
#pragma unroll
    for (int r = 0; r < 8; ++r) {
      float p0 = kill[0] ? 0.f : __expf(sc[0][r] - mrow[r]);
      float p1 = kill[1] ? 0.f : __expf(sc[1][r] - mrow[r]);
      sc[0][r] = p0;
      sc[1][r] = p1;
      float rv = p0 + p1;
#pragma unroll
      for (int m = 8; m >= 1; m >>= 1) rv += __shfl_xor(rv, m, 16);
      lrow[r] = lrow[r] * alpha[r] + rv;
    }
#pragma unroll
    for (int t = 0; t < 4; ++t)
#pragma unroll
      for (int r = 0; r < 8; ++r) cacc[t][r] *= alpha[r];

    // P: C layout -> A layout via per-wave LDS bounce (in-order within wave)
    const int prow = (lane < 16) ? 0 : 8;
#pragma unroll
    for (int u = 0; u < 2; ++u)
#pragma unroll
      for (int r = 0; r < 8; ++r)
        Pb[wave][prow + r][u * 16 + lm] = (__bf16)sc[u][r];

    v16bf pf = ld_frag_a(&Pb[wave][lm][0], khalf);
#pragma unroll
    for (int t = 0; t < 4; ++t)
      cacc[t] = wmma_bf16(pf, ld_frag_b(&Vt[buf][t * 16 + lm][0], khalf),
                          cacc[t]);
  }

  // normalize and scatter ctx back to [B,S,Dm] bf16
  const int row_off = (lane < 16) ? 0 : 8;
#pragma unroll
  for (int t = 0; t < 4; ++t)
#pragma unroll
    for (int r = 0; r < 8; ++r) {
      int srow = qrow0 + row_off + r;
      int col = h * 64 + t * 16 + lm;
      float denom = lrow[r];
      float v = (denom > 0.f) ? (cacc[t][r] / denom) : 0.f;
      ctx[((size_t)(b * Sn + srow)) * Dm + col] = (__bf16)v;
    }
}

// ---------------------------------------------------------------------------
// SwiGLU: act = h0 * silu(h1)
// ---------------------------------------------------------------------------
__global__ void swiglu_kernel(const __bf16* __restrict__ hbuf,
                              __bf16* __restrict__ act, long long rows,
                              int inter) {
  long long idx = (long long)blockIdx.x * blockDim.x + threadIdx.x;
  long long tot = rows * inter;
  if (idx >= tot) return;
  long long r = idx / inter;
  int j = (int)(idx % inter);
  float a = (float)hbuf[(size_t)r * (2 * inter) + j];
  float g = (float)hbuf[(size_t)r * (2 * inter) + inter + j];
  float sg = g / (1.f + __expf(-g));
  act[(size_t)r * inter + j] = (__bf16)(a * sg);
}

// ---------------------------------------------------------------------------
extern "C" void kernel_launch(void* const* d_in, const int* in_sizes, int n_in,
                              void* d_out, int out_size, void* d_ws,
                              size_t ws_size, hipStream_t stream) {
  (void)in_sizes; (void)n_in; (void)out_size; (void)ws_size;
  constexpr int Bq = 2, Sq = 2048, Dm = 1024, Hn = 16, INTER = 4096;
  constexpr int Mr = Bq * Sq;  // 4096 rows

  const float* hid   = (const float*)d_in[0];
  const float* mask  = (const float*)d_in[1];
  const float* freqs = (const float*)d_in[2];
  const float* Wq = (const float*)d_in[3];  const float* bq = (const float*)d_in[4];
  const float* Wk = (const float*)d_in[5];  const float* bk = (const float*)d_in[6];
  const float* Wv = (const float*)d_in[7];  const float* bv = (const float*)d_in[8];
  const float* Wo = (const float*)d_in[9];  const float* bo = (const float*)d_in[10];
  const float* g1 = (const float*)d_in[11]; const float* be1 = (const float*)d_in[12];
  const float* g2 = (const float*)d_in[13]; const float* be2 = (const float*)d_in[14];
  const float* W1 = (const float*)d_in[15]; const float* b1 = (const float*)d_in[16];
  const float* W2 = (const float*)d_in[17]; const float* b2 = (const float*)d_in[18];

  char* ws = (char*)d_ws;
  size_t off = 0;
  auto take = [&](size_t bytes) -> char* {
    char* p = ws + off;
    off += (bytes + 255) & ~(size_t)255;
    return p;
  };
  __bf16* wqb = (__bf16*)take((size_t)Dm * Dm * 2);
  __bf16* wkb = (__bf16*)take((size_t)Dm * Dm * 2);
  __bf16* wvb = (__bf16*)take((size_t)Dm * Dm * 2);
  __bf16* wob = (__bf16*)take((size_t)Dm * Dm * 2);
  __bf16* w1b = (__bf16*)take((size_t)Dm * 2 * INTER * 2);
  __bf16* w2b = (__bf16*)take((size_t)INTER * Dm * 2);
  __bf16* xln = (__bf16*)take((size_t)Mr * Dm * 2);
  float*  Qf  = (float*)take((size_t)Mr * Dm * 4);
  float*  Kf  = (float*)take((size_t)Mr * Dm * 4);
  float*  Vf  = (float*)take((size_t)Mr * Dm * 4);
  __bf16* Qbb = (__bf16*)take((size_t)Mr * Dm * 2);
  __bf16* Kbb = (__bf16*)take((size_t)Mr * Dm * 2);
  __bf16* Vbb = (__bf16*)take((size_t)Mr * Dm * 2);
  __bf16* ctx = (__bf16*)take((size_t)Mr * Dm * 2);
  float*  hiddenf = (float*)take((size_t)Mr * Dm * 4);
  __bf16* ybuf = (__bf16*)take((size_t)Mr * Dm * 2);
  __bf16* hbuf = (__bf16*)take((size_t)Mr * 2 * INTER * 2);
  __bf16* actb = (__bf16*)take((size_t)Mr * INTER * 2);

  // 1. weights -> bf16
  cvt_f32_bf16<<<2048, 256, 0, stream>>>(Wq, wqb, (long long)Dm * Dm);
  cvt_f32_bf16<<<2048, 256, 0, stream>>>(Wk, wkb, (long long)Dm * Dm);
  cvt_f32_bf16<<<2048, 256, 0, stream>>>(Wv, wvb, (long long)Dm * Dm);
  cvt_f32_bf16<<<2048, 256, 0, stream>>>(Wo, wob, (long long)Dm * Dm);
  cvt_f32_bf16<<<4096, 256, 0, stream>>>(W1, w1b, (long long)Dm * 2 * INTER);
  cvt_f32_bf16<<<4096, 256, 0, stream>>>(W2, w2b, (long long)INTER * Dm);

  // 2. LN1
  ln_kernel<<<Mr, 256, 0, stream>>>(hid, g1, be1, xln, Dm);

  // 3. QKV projections
  dim3 g_sq(Dm / 128, Mr / 128);
  gemm_bf16_kernel<false, false><<<g_sq, 256, 0, stream>>>(
      xln, wqb, bq, nullptr, Qf, nullptr, Mr, Dm, Dm);
  gemm_bf16_kernel<false, false><<<g_sq, 256, 0, stream>>>(
      xln, wkb, bk, nullptr, Kf, nullptr, Mr, Dm, Dm);
  gemm_bf16_kernel<false, false><<<g_sq, 256, 0, stream>>>(
      xln, wvb, bv, nullptr, Vf, nullptr, Mr, Dm, Dm);

  // 4. RoPE + head-major bf16 repack
  int ropeN = Bq * Sq * Hn * 32;
  rope_pack_kernel<<<(ropeN + 255) / 256, 256, 0, stream>>>(
      Qf, Kf, Vf, freqs, Qbb, Kbb, Vbb, Bq, Hn, Sq, Dm);

  // 5. flash attention
  dim3 g_at(Sq / 128, Bq * Hn);
  attn_kernel<<<g_at, 256, 0, stream>>>(Qbb, Kbb, Vbb, mask, ctx, Hn, Sq, Dm);

  // 6. O projection + residual
  gemm_bf16_kernel<false, true><<<g_sq, 256, 0, stream>>>(
      ctx, wob, bo, hid, hiddenf, nullptr, Mr, Dm, Dm);

  // 7. LN2
  ln_kernel<<<Mr, 256, 0, stream>>>(hiddenf, g2, be2, ybuf, Dm);

  // 8. W1 (gated up-proj, bf16 out)
  dim3 g_w1((2 * INTER) / 128, Mr / 128);
  gemm_bf16_kernel<true, false><<<g_w1, 256, 0, stream>>>(
      ybuf, w1b, b1, nullptr, nullptr, hbuf, Mr, 2 * INTER, Dm);

  // 9. SwiGLU
  long long swN = (long long)Mr * INTER;
  swiglu_kernel<<<(unsigned)((swN + 255) / 256), 256, 0, stream>>>(hbuf, actb,
                                                                   Mr, INTER);

  // 10. W2 + residual -> final output (fp32)
  dim3 g_w2(Dm / 128, Mr / 128);
  gemm_bf16_kernel<false, true><<<g_w2, 256, 0, stream>>>(
      actb, w2b, b2, hiddenf, (float*)d_out, nullptr, Mr, Dm, INTER);
}
